// Block_7559142441593
// MI455X (gfx1250) — compile-verified, matrix operand order checked by reading, not run on hardware
//
#include <hip/hip_runtime.h>
#include <cmath>
#include <cstdint>
#include <cstddef>

// ---------------------------------------------------------------------------
// Types / helpers for CDNA5 (gfx1250) wave32 WMMA
// ---------------------------------------------------------------------------
typedef __attribute__((ext_vector_type(16))) __bf16 v16bf;
typedef __attribute__((ext_vector_type(8)))  float  v8f;

#define LDK 48  // LDS K-stride (elements); 48*2=96B, multiple of 16B for uint4 ops

__device__ __forceinline__ float gelu_exact(float x) {
  return 0.5f * x * (1.0f + erff(x * 0.70710678118654752440f));
}

__device__ __forceinline__ v8f wmma_bf16(v16bf a, v16bf b, v8f c) {
  // D = A(16x32 bf16) * B(32x16 bf16) + C(16x16 f32)
  return __builtin_amdgcn_wmma_f32_16x16x32_bf16(false, a, false, b,
                                                 (short)0, c, false, false);
}

// Load an A/B fragment from row-major LDS: lane holds elements
// K = kb..kb+7 and kb+16..kb+23 of its row (ISA 7.12.2 16-bit layout).
__device__ __forceinline__ v16bf frag_ld(const __bf16* rowbase, int kb) {
  union { uint4 u; __bf16 h[8]; } lo, hi;
  lo.u = *(const uint4*)(rowbase + kb);
  hi.u = *(const uint4*)(rowbase + kb + 16);
  v16bf r;
#pragma unroll
  for (int j = 0; j < 8; ++j) { r[j] = lo.h[j]; r[8 + j] = hi.h[j]; }
  return r;
}

// Stage 16 contiguous elements (bf16 pass-through or f32->bf16) into LDS.
template<typename AT>
__device__ __forceinline__ void stage_a16(const AT* src, __bf16* dst) {
  if constexpr (sizeof(AT) == 2) {
    ((uint4*)dst)[0] = ((const uint4*)src)[0];
    ((uint4*)dst)[1] = ((const uint4*)src)[1];
  } else {
    const float4* s4 = (const float4*)src;
    float4 f[4];
#pragma unroll
    for (int i = 0; i < 4; ++i) f[i] = s4[i];
    const float* ff = (const float*)f;
    union { uint4 u; __bf16 h[8]; } p0, p1;
#pragma unroll
    for (int j = 0; j < 8; ++j) { p0.h[j] = (__bf16)ff[j]; p1.h[j] = (__bf16)ff[8 + j]; }
    ((uint4*)dst)[0] = p0.u;
    ((uint4*)dst)[1] = p1.u;
  }
}

__device__ __forceinline__ float block_sum256(float v, float* red) {
  const int tid = threadIdx.x;
  red[tid] = v;
  __syncthreads();
#pragma unroll
  for (int off = 128; off > 0; off >>= 1) {
    if (tid < off) red[tid] += red[tid + off];
    __syncthreads();
  }
  const float r = red[0];
  __syncthreads();
  return r;
}

// ---------------------------------------------------------------------------
// Generic bf16 WMMA GEMM: out[M,N] = act(A[M,K] * B + bias) (+residual)
//   BL=0: B is weight[N,K] (torch linear, x @ w.T);  BL=1: B is weight[K,N]
//   MODE: 0 = bf16 out, 1 = bf16 out + exact GELU, 2 = f32 out = resid + val
// Block tile 128x64, 8 waves (4x2), each wave 32x32 (4 wmma / K-step of 32).
// ---------------------------------------------------------------------------
template<typename AT, int BL, int MODE>
__global__ __launch_bounds__(256) void gemm_wmma_kernel(
    const AT* __restrict__ A, const float* __restrict__ Bw,
    const float* __restrict__ bias, const float* __restrict__ resid,
    void* __restrict__ outv, int M, int N, int K)
{
  __shared__ alignas(16) __bf16 As[128][LDK];
  __shared__ alignas(16) __bf16 Bs[64][LDK];
  const int tid = threadIdx.x, lane = tid & 31, wave = tid >> 5;
  const int wr = wave & 3, wc = wave >> 2;
  const int bn0 = blockIdx.x * 64;
  const int bm0 = blockIdx.y * 128;
  const v8f vz = {0.f,0.f,0.f,0.f,0.f,0.f,0.f,0.f};
  v8f acc[2][2] = {{vz, vz}, {vz, vz}};
  const int arow = tid >> 1, akh = (tid & 1) * 16;

  for (int k0 = 0; k0 < K; k0 += 32) {
    stage_a16<AT>(A + (size_t)(bm0 + arow) * K + k0 + akh, &As[arow][akh]);
    if (BL == 0) {
      const int n = tid >> 2, kq = (tid & 3) * 8;
      const float* src = Bw + (size_t)(bn0 + n) * K + k0 + kq;
      union { uint4 u; __bf16 h[8]; } pk;
#pragma unroll
      for (int j = 0; j < 8; ++j) pk.h[j] = (__bf16)src[j];
      *(uint4*)&Bs[n][kq] = pk.u;
    } else {
      const int n = tid & 63, ks = (tid >> 6) * 8;
      union { uint4 u; __bf16 h[8]; } pk;
#pragma unroll
      for (int j = 0; j < 8; ++j) pk.h[j] = (__bf16)Bw[(size_t)(k0 + ks + j) * N + bn0 + n];
      *(uint4*)&Bs[n][ks] = pk.u;
    }
    __syncthreads();
    const int fr = lane & 15, kb = (lane >> 4) * 8;
    v16bf a0 = frag_ld(&As[wr * 32 + fr][0], kb);
    v16bf a1 = frag_ld(&As[wr * 32 + 16 + fr][0], kb);
    v16bf b0 = frag_ld(&Bs[wc * 32 + fr][0], kb);
    v16bf b1 = frag_ld(&Bs[wc * 32 + 16 + fr][0], kb);
    acc[0][0] = wmma_bf16(a0, b0, acc[0][0]);
    acc[0][1] = wmma_bf16(a0, b1, acc[0][1]);
    acc[1][0] = wmma_bf16(a1, b0, acc[1][0]);
    acc[1][1] = wmma_bf16(a1, b1, acc[1][1]);
    __syncthreads();
  }

#pragma unroll
  for (int ti = 0; ti < 2; ++ti)
#pragma unroll
    for (int tj = 0; tj < 2; ++tj)
#pragma unroll
      for (int v = 0; v < 8; ++v) {
        const int m = bm0 + wr * 32 + ti * 16 + ((lane >> 4) * 8) + v;
        const int n = bn0 + wc * 32 + tj * 16 + (lane & 15);
        if (m < M && n < N) {
          float val = acc[ti][tj][v] + (bias ? bias[n] : 0.f);
          if (MODE == 1) val = gelu_exact(val);
          if (MODE == 2)
            ((float*)outv)[(size_t)m * N + n] = resid[(size_t)m * N + n] + val;
          else
            ((__bf16*)outv)[(size_t)m * N + n] = (__bf16)val;
        }
      }
}

// ---------------------------------------------------------------------------
// 3x3 SAME conv as implicit GEMM (9 taps x 1x1 WMMA GEMM), C=384 -> 384.
// X: bf16 [8,56,56,384] row-ordered as [M=25088, 384]; W: f32 [3,3,384,384].
// ---------------------------------------------------------------------------
__global__ __launch_bounds__(256) void conv3x3_wmma_kernel(
    const __bf16* __restrict__ X, const float* __restrict__ W,
    __bf16* __restrict__ out)
{
  __shared__ alignas(16) __bf16 As[128][LDK];
  __shared__ alignas(16) __bf16 Bs[64][LDK];
  const int tid = threadIdx.x, lane = tid & 31, wave = tid >> 5;
  const int wr = wave & 3, wc = wave >> 2;
  const int bn0 = blockIdx.x * 64;
  const int bm0 = blockIdx.y * 128;
  const v8f vz = {0.f,0.f,0.f,0.f,0.f,0.f,0.f,0.f};
  v8f acc[2][2] = {{vz, vz}, {vz, vz}};
  const int arow = tid >> 1, akh = (tid & 1) * 16;
  const int m = bm0 + arow;
  const int bb = m / 3136, hw = m % 3136, yy = hw / 56, xx = hw % 56;

  for (int tap = 0; tap < 9; ++tap) {
    const int ky = tap / 3 - 1, kx = tap % 3 - 1;
    const int y = yy + ky, x = xx + kx;
    const bool valid = ((unsigned)y < 56u) && ((unsigned)x < 56u);
    const __bf16* asrc =
        X + (((size_t)bb * 56 + (valid ? y : 0)) * 56 + (valid ? x : 0)) * 384;
    const float* wtap = W + (size_t)tap * 384 * 384;
    for (int k0 = 0; k0 < 384; k0 += 32) {
      if (valid) {
        stage_a16<__bf16>(asrc + k0 + akh, &As[arow][akh]);
      } else {
        uint4 z; z.x = z.y = z.z = z.w = 0u;
        ((uint4*)&As[arow][akh])[0] = z;
        ((uint4*)&As[arow][akh])[1] = z;
      }
      const int n = tid & 63, ks = (tid >> 6) * 8;
      union { uint4 u; __bf16 h[8]; } pk;
#pragma unroll
      for (int j = 0; j < 8; ++j) pk.h[j] = (__bf16)wtap[(size_t)(k0 + ks + j) * 384 + bn0 + n];
      *(uint4*)&Bs[n][ks] = pk.u;
      __syncthreads();
      const int fr = lane & 15, kb = (lane >> 4) * 8;
      v16bf a0 = frag_ld(&As[wr * 32 + fr][0], kb);
      v16bf a1 = frag_ld(&As[wr * 32 + 16 + fr][0], kb);
      v16bf b0 = frag_ld(&Bs[wc * 32 + fr][0], kb);
      v16bf b1 = frag_ld(&Bs[wc * 32 + 16 + fr][0], kb);
      acc[0][0] = wmma_bf16(a0, b0, acc[0][0]);
      acc[0][1] = wmma_bf16(a0, b1, acc[0][1]);
      acc[1][0] = wmma_bf16(a1, b0, acc[1][0]);
      acc[1][1] = wmma_bf16(a1, b1, acc[1][1]);
      __syncthreads();
    }
  }

#pragma unroll
  for (int ti = 0; ti < 2; ++ti)
#pragma unroll
    for (int tj = 0; tj < 2; ++tj)
#pragma unroll
      for (int v = 0; v < 8; ++v) {
        const int mm = bm0 + wr * 32 + ti * 16 + ((lane >> 4) * 8) + v;
        const int nn = bn0 + wc * 32 + tj * 16 + (lane & 15);
        if (mm < 25088 && nn < 384)
          out[(size_t)mm * 384 + nn] = (__bf16)acc[ti][tj][v];
      }
}

// ---------------------------------------------------------------------------
// Windowed attention, one block per (32 Q rows, head, window). Two-pass
// softmax with S held in LDS (bf16), rel-pos bias decomposed per row.
// qkv: bf16 [128*196, 2304] (cols: 3 x 12 heads x 64); out: bf16 [128*196,768]
// ---------------------------------------------------------------------------
__global__ __launch_bounds__(256) void attn_kernel(
    const __bf16* __restrict__ qkv, const float* __restrict__ rph,
    const float* __restrict__ rpw, __bf16* __restrict__ out)
{
  const int rb  = blockIdx.x;  // 0..6 : 32-row blocks (224 padded rows)
  const int hh  = blockIdx.y;  // 0..11
  const int wdx = blockIdx.z;  // 0..127
  const int tid = threadIdx.x, lane = tid & 31, wave = tid >> 5;

  __shared__ alignas(16) __bf16 Qs[32][72];
  __shared__ alignas(16) __bf16 KVs[224][72];
  __shared__ alignas(16) __bf16 SP[32][224];
  __shared__ float relh[32][16];
  __shared__ float relw[32][16];

  const size_t base = (size_t)wdx * 196 * 2304;
  const __bf16* qb = qkv + base + hh * 64;
  const __bf16* kp = qkv + base + 768 + hh * 64;
  const __bf16* vp = qkv + base + 1536 + hh * 64;
  uint4 z4; z4.x = z4.y = z4.z = z4.w = 0u;

  // stage Q (32 rows x 64 d, unscaled) and K (224 rows x 64 d)
  for (int idx = tid; idx < 32 * 8; idx += 256) {
    const int r = idx >> 3, dq = (idx & 7) * 8;
    const int m = rb * 32 + r;
    uint4 u = (m < 196) ? *(const uint4*)(qb + (size_t)m * 2304 + dq) : z4;
    *(uint4*)&Qs[r][dq] = u;
  }
  for (int idx = tid; idx < 224 * 8; idx += 256) {
    const int r = idx >> 3, dq = (idx & 7) * 8;
    uint4 u = (r < 196) ? *(const uint4*)(kp + (size_t)r * 2304 + dq) : z4;
    *(uint4*)&KVs[r][dq] = u;
  }
  __syncthreads();

  // decomposed relative-position bias: relh[r][p'] , relw[r][q']
  for (int idx = tid; idx < 32 * 28; idx += 256) {
    const int r = idx / 28, j = idx % 28;
    const int m = rb * 32 + r;
    const int mc = (m < 196) ? m : 0;
    const int p = mc / 14, q = mc % 14;
    const float* tab = (j < 14) ? (rph + (size_t)(p - j + 13) * 64)
                                : (rpw + (size_t)(q - (j - 14) + 13) * 64);
    float s = 0.f;
#pragma unroll 8
    for (int d = 0; d < 64; ++d) s += (float)Qs[r][d] * tab[d];
    if (j < 14) relh[r][j] = s; else relw[r][j - 14] = s;
  }
  __syncthreads();

  // phase 1: S = 0.125 * Q K^T + bias, masked to -1e30 beyond col 195
  for (int t = wave; t < 28; t += 8) {
    const int ti = t / 14, tj = t % 14;
    v8f acc = {0.f,0.f,0.f,0.f,0.f,0.f,0.f,0.f};
    const int fr = lane & 15;
#pragma unroll
    for (int ch = 0; ch < 2; ++ch) {
      const int kb = ch * 32 + (lane >> 4) * 8;
      v16bf a = frag_ld(&Qs[ti * 16 + fr][0], kb);
      v16bf b = frag_ld(&KVs[tj * 16 + fr][0], kb);
      acc = wmma_bf16(a, b, acc);
    }
#pragma unroll
    for (int v = 0; v < 8; ++v) {
      const int r = ti * 16 + (lane >> 4) * 8 + v;
      const int gc = tj * 16 + (lane & 15);
      float val = (gc < 196)
          ? acc[v] * 0.125f + relh[r][gc / 14] + relw[r][gc % 14]
          : -1e30f;
      SP[r][gc] = (__bf16)val;
    }
  }
  __syncthreads();

  // stage V (overwrites K region; phase-1 reads done)
  for (int idx = tid; idx < 224 * 8; idx += 256) {
    const int r = idx >> 3, dq = (idx & 7) * 8;
    uint4 u = (r < 196) ? *(const uint4*)(vp + (size_t)r * 2304 + dq) : z4;
    *(uint4*)&KVs[r][dq] = u;
  }
  // softmax, one thread per row (in-place, same-row only)
  if (tid < 32) {
    const int r = tid;
    float mx = -3e30f;
    for (int c = 0; c < 224; ++c) mx = fmaxf(mx, (float)SP[r][c]);
    float sum = 0.f;
    for (int c = 0; c < 224; ++c) {
      const float e = expf((float)SP[r][c] - mx);
      sum += e;
      SP[r][c] = (__bf16)e;
    }
    const float inv = 1.0f / sum;
    for (int c = 0; c < 224; ++c) SP[r][c] = (__bf16)((float)SP[r][c] * inv);
  }
  __syncthreads();

  // phase 2: O = P V ; wave -> (row tile tm 0..1, d tile tn 0..3)
  {
    const int tm = wave & 1, tn = wave >> 1;
    v8f acc = {0.f,0.f,0.f,0.f,0.f,0.f,0.f,0.f};
    const int fr = lane & 15;
    const int kb = (lane >> 4) * 8;
    const int col = tn * 16 + fr;
#pragma unroll
    for (int ch = 0; ch < 7; ++ch) {
      v16bf a = frag_ld(&SP[tm * 16 + fr][0], ch * 32 + kb);
      v16bf b;
#pragma unroll
      for (int j = 0; j < 8; ++j) {
        b[j]     = KVs[ch * 32 + kb + j][col];
        b[8 + j] = KVs[ch * 32 + kb + 16 + j][col];
      }
      acc = wmma_bf16(a, b, acc);
    }
#pragma unroll
    for (int v = 0; v < 8; ++v) {
      const int r = tm * 16 + (lane >> 4) * 8 + v;
      const int m = rb * 32 + r;
      if (m < 196)
        out[((size_t)wdx * 196 + m) * 768 + hh * 64 + tn * 16 + (lane & 15)]
            = (__bf16)acc[v];
    }
  }
}

// ---------------------------------------------------------------------------
// LayerNorm(norm1) fused with window partition: x[8,56,56,768] -> bf16[25088,768]
// windowed row r = (b*16 + wy*4 + wx)*196 + py*14 + px
// ---------------------------------------------------------------------------
__global__ __launch_bounds__(256) void ln_partition_kernel(
    const float* __restrict__ x, const float* __restrict__ g,
    const float* __restrict__ be, __bf16* __restrict__ out)
{
  __shared__ float red[256];
  const int tid = threadIdx.x;
  const int r = blockIdx.x;
  const int w = r / 196, n = r % 196;
  const int b = w >> 4, wy = (w >> 2) & 3, wx = w & 3;
  const int y = wy * 14 + n / 14, xx = wx * 14 + n % 14;
  const float* src = x + (((size_t)b * 56 + y) * 56 + xx) * 768;
  float vals[3];
  float s = 0.f, s2 = 0.f;
#pragma unroll
  for (int i = 0; i < 3; ++i) {
    vals[i] = src[tid + i * 256];
    s += vals[i]; s2 += vals[i] * vals[i];
  }
  s  = block_sum256(s, red);
  s2 = block_sum256(s2, red);
  const float mean = s * (1.f / 768.f);
  const float var  = s2 * (1.f / 768.f) - mean * mean;
  const float inv  = rsqrtf(var + 1e-5f);
#pragma unroll
  for (int i = 0; i < 3; ++i) {
    const int c = tid + i * 256;
    out[(size_t)r * 768 + c] = (__bf16)((vals[i] - mean) * inv * g[c] + be[c]);
  }
}

// Generic per-row LayerNorm (+optional GELU, +optional f32 residual add)
template<typename TIN, typename TOUT, bool GELU, bool ADDRES>
__global__ __launch_bounds__(256) void ln_rows_kernel(
    const TIN* __restrict__ in, const float* __restrict__ g,
    const float* __restrict__ be, float eps, TOUT* __restrict__ out,
    const float* __restrict__ resid, int D)
{
  __shared__ float red[256];
  const int tid = threadIdx.x;
  const size_t r = blockIdx.x;
  const TIN* src = in + r * D;
  float s = 0.f, s2 = 0.f;
  for (int c = tid; c < D; c += 256) { const float v = (float)src[c]; s += v; s2 += v * v; }
  s  = block_sum256(s, red);
  s2 = block_sum256(s2, red);
  const float mean = s / (float)D;
  const float var  = s2 / (float)D - mean * mean;
  const float inv  = rsqrtf(var + eps);
  for (int c = tid; c < D; c += 256) {
    float v = ((float)src[c] - mean) * inv * g[c] + be[c];
    if (GELU) v = gelu_exact(v);
    float o = v;
    if (ADDRES) o += resid[r * D + c];
    out[r * D + c] = (TOUT)o;
  }
}

// x1[B,HW,C] = x + window_unpartition(projw)
__global__ __launch_bounds__(256) void unpart_add_kernel(
    const float* __restrict__ x, const __bf16* __restrict__ pw,
    float* __restrict__ x1)
{
  const size_t idx = (size_t)blockIdx.x * 256 + threadIdx.x;
  if (idx >= (size_t)25088 * 768) return;
  const int c = (int)(idx % 768);
  const size_t rr = idx / 768;
  const int b = (int)(rr / 3136);
  const int hw = (int)(rr % 3136);
  const int y = hw / 56, xx = hw % 56;
  const int w = (b * 4 + y / 14) * 4 + xx / 14;
  const int n = (y % 14) * 14 + (xx % 14);
  x1[idx] = x[idx] + (float)pw[((size_t)w * 196 + n) * 768 + c];
}

// ---------------------------------------------------------------------------
// Host launcher
// ---------------------------------------------------------------------------
extern "C" void kernel_launch(void* const* d_in, const int* in_sizes, int n_in,
                              void* d_out, int out_size, void* d_ws, size_t ws_size,
                              hipStream_t stream)
{
  (void)in_sizes; (void)n_in; (void)out_size; (void)ws_size;
  const float* x       = (const float*)d_in[0];
  const float* norm1_w = (const float*)d_in[1];
  const float* norm1_b = (const float*)d_in[2];
  const float* qkv_w   = (const float*)d_in[3];
  const float* qkv_b   = (const float*)d_in[4];
  const float* rph     = (const float*)d_in[5];
  const float* rpw     = (const float*)d_in[6];
  const float* proj_w  = (const float*)d_in[7];
  const float* proj_b  = (const float*)d_in[8];
  const float* norm2_w = (const float*)d_in[9];
  const float* norm2_b = (const float*)d_in[10];
  const float* fc1_w   = (const float*)d_in[11];
  const float* fc1_b   = (const float*)d_in[12];
  const float* fc2_w   = (const float*)d_in[13];
  const float* fc2_b   = (const float*)d_in[14];
  const float* conv1_w = (const float*)d_in[15];
  const float* ln1_w   = (const float*)d_in[16];
  const float* ln1_b   = (const float*)d_in[17];
  const float* conv2_w = (const float*)d_in[18];
  const float* ln2c_w  = (const float*)d_in[19];
  const float* ln2c_b  = (const float*)d_in[20];
  const float* conv3_w = (const float*)d_in[21];
  const float* ln3_w   = (const float*)d_in[22];
  const float* ln3_b   = (const float*)d_in[23];

  const size_t M = 25088;  // 8*56*56 tokens == 128 windows * 196
  char* ws = (char*)d_ws;
  auto alup = [](size_t v) { return (v + 255) & ~(size_t)255; };
  const size_t offA = 0;                 const size_t szA = alup(M * 3072 * 2); // qkv / fc1 (bf16)
  const size_t offB = offA + szA;        const size_t szB = alup(M * 768 * 2);  // xw / attnout / h2
  const size_t offC = offB + szB;        const size_t szC = alup(M * 768 * 2);  // projw ; c1a | c2a
  const size_t offD = offC + szC;        const size_t szD = alup(M * 768 * 4);  // x1 (f32)
  const size_t offE = offD + szD;        const size_t szE = alup(M * 768 * 4);  // ori (f32)
  const size_t offF = offE + szE;                                               // c1buf|c2buf -> c3buf

  __bf16* qkvbuf  = (__bf16*)(ws + offA);
  __bf16* fc1buf  = (__bf16*)(ws + offA);
  __bf16* xw      = (__bf16*)(ws + offB);
  __bf16* attnout = (__bf16*)(ws + offB);
  __bf16* h2      = (__bf16*)(ws + offB);
  __bf16* projw   = (__bf16*)(ws + offC);
  __bf16* c1a     = (__bf16*)(ws + offC);
  __bf16* c2a     = (__bf16*)(ws + offC) + M * 384;
  float*  x1      = (float*)(ws + offD);
  float*  ori     = (float*)(ws + offE);
  __bf16* c1buf   = (__bf16*)(ws + offF);
  __bf16* c2buf   = (__bf16*)(ws + offF) + M * 384;
  __bf16* c3buf   = (__bf16*)(ws + offF);
  float*  outp    = (float*)d_out;

  const dim3 blk(256);
  const int MT = (int)(M / 128);  // 196 row tiles

  // 1) LN(norm1) + window partition
  ln_partition_kernel<<<(int)M, blk, 0, stream>>>(x, norm1_w, norm1_b, xw);
  // 2) qkv = xw @ qkv_w.T + qkv_b
  gemm_wmma_kernel<__bf16, 0, 0><<<dim3(2304 / 64, MT), blk, 0, stream>>>(
      xw, qkv_w, qkv_b, nullptr, qkvbuf, (int)M, 2304, 768);
  // 3) windowed attention with decomposed rel-pos bias
  attn_kernel<<<dim3(7, 12, 128), blk, 0, stream>>>(qkvbuf, rph, rpw, attnout);
  // 4) proj
  gemm_wmma_kernel<__bf16, 0, 0><<<dim3(768 / 64, MT), blk, 0, stream>>>(
      attnout, proj_w, proj_b, nullptr, projw, (int)M, 768, 768);
  // 5) x1 = shortcut + unpartition(proj)
  unpart_add_kernel<<<(int)((M * 768 + 255) / 256), blk, 0, stream>>>(x, projw, x1);
  // 6) LN(norm2)
  ln_rows_kernel<float, __bf16, false, false><<<(int)M, blk, 0, stream>>>(
      x1, norm2_w, norm2_b, 1e-5f, h2, nullptr, 768);
  // 7) fc1 + GELU
  gemm_wmma_kernel<__bf16, 0, 1><<<dim3(3072 / 64, MT), blk, 0, stream>>>(
      h2, fc1_w, fc1_b, nullptr, fc1buf, (int)M, 3072, 768);
  // 8) ori = x1 + fc2(...)
  gemm_wmma_kernel<__bf16, 0, 2><<<dim3(768 / 64, MT), blk, 0, stream>>>(
      fc1buf, fc2_w, fc2_b, x1, ori, (int)M, 768, 3072);
  // 9) conv1 (1x1, 768->384), no bias
  gemm_wmma_kernel<float, 1, 0><<<dim3(384 / 64, MT), blk, 0, stream>>>(
      ori, conv1_w, nullptr, nullptr, c1buf, (int)M, 384, 768);
  // 10) LN(ln1, 1e-6) + GELU
  ln_rows_kernel<__bf16, __bf16, true, false><<<(int)M, blk, 0, stream>>>(
      c1buf, ln1_w, ln1_b, 1e-6f, c1a, nullptr, 384);
  // 11) conv2 (3x3 SAME, 384->384)
  conv3x3_wmma_kernel<<<dim3(384 / 64, MT), blk, 0, stream>>>(c1a, conv2_w, c2buf);
  // 12) LN(ln2, 1e-6) + GELU
  ln_rows_kernel<__bf16, __bf16, true, false><<<(int)M, blk, 0, stream>>>(
      c2buf, ln2c_w, ln2c_b, 1e-6f, c2a, nullptr, 384);
  // 13) conv3 (1x1, 384->768)
  gemm_wmma_kernel<__bf16, 1, 0><<<dim3(768 / 64, MT), blk, 0, stream>>>(
      c2a, conv3_w, nullptr, nullptr, c3buf, (int)M, 768, 384);
  // 14) out = ori + LN(ln3, 1e-6)
  ln_rows_kernel<__bf16, float, false, true><<<(int)M, blk, 0, stream>>>(
      c3buf, ln3_w, ln3_b, 1e-6f, outp, ori, 768);
}
